// RecurrentNeuralNetwork_41214506172657
// MI455X (gfx1250) — matrix operand order repaired and checked by reading, not compiled, not run on hardware
//
#include <hip/hip_runtime.h>

// Problem sizes (fixed by the reference)
#define T_STEPS 16384
#define XD      256
#define HD      1024

// Scan decomposition: 64 persistent workgroups x 256 threads (8 waves).
#define NWG      64
#define SCAN_TPB 256

typedef _Float16 v16h __attribute__((ext_vector_type(16)));
typedef float    v8f  __attribute__((ext_vector_type(8)));

#if defined(__HIP_DEVICE_COMPILE__)
#if !__has_builtin(__builtin_amdgcn_wmma_f32_16x16x32_f16)
#error "wmma_f32_16x16x32_f16 builtin not available on device"
#endif
#endif

// ---------------------------------------------------------------------------
// Zero the cross-workgroup step flags (d_ws is poisoned with 0xAA by harness).
// ---------------------------------------------------------------------------
__global__ void rnn_init_flags(unsigned int* flags) {
  if (threadIdx.x < NWG) flags[threadIdx.x] = 0u;
}

// ---------------------------------------------------------------------------
// Bx = x_seq @ B^T + c, written into d_out (scan overwrites it in place).
// One wave per 16x16 output tile, f16 WMMA (f32 accumulate), K in steps of 32.
//   A (16x32 f16), lane L: M=L%16, elems 0-15 = K=k0+off..,
//       off = (L<16?0:16)   (contiguous K order, same as B)
//   B (32x16 f16), lane L: N=L%16, elems 0-15 = B^T[k0+(L<16?0:16)..][N]
//       = contiguous B[n][k] since B is row-major (H,X)
//   D (16x16 f32): VGPR v, lanes 0-15 -> M=v, lanes 16-31 -> M=v+8, N=L%16.
// ---------------------------------------------------------------------------
__global__ void rnn_gemm_bx(const float* __restrict__ x,
                            const float* __restrict__ Bm,
                            const float* __restrict__ c,
                            float* __restrict__ out) {
  const int lane = threadIdx.x & 31;
  const int wave = threadIdx.x >> 5;
  const int trow = blockIdx.x;               // T/16 = 1024 tile rows
  const int tcol = blockIdx.y * 8 + wave;    // H/16 = 64 tile cols, 8 per block
  const int mr   = lane & 15;
  const int ahl  = (lane < 16) ? 0 : 16;     // A-operand K sub-offset
  const int bhl  = (lane < 16) ? 0 : 16;     // B-operand K sub-offset

  const float* xrow = x  + (size_t)(trow * 16 + mr) * XD;
  const float* brow = Bm + (size_t)(tcol * 16 + mr) * XD;

  const float cv = c[tcol * 16 + mr];
  v8f acc = {cv, cv, cv, cv, cv, cv, cv, cv};

#pragma unroll
  for (int k0 = 0; k0 < XD; k0 += 32) {
    const float* ap = xrow + k0 + ahl;
    const float* bp = brow + k0 + bhl;
    v16h a, b;
#pragma unroll
    for (int i = 0; i < 8; ++i) {
      a[i]     = (_Float16)ap[i];        // K = k0+off .. +7
      a[i + 8] = (_Float16)ap[i + 8];    // K = k0+off+8 .. +15
    }
#pragma unroll
    for (int i = 0; i < 16; ++i)
      b[i] = (_Float16)bp[i];            // K = k0+off .. +15 (of B^T column N)

#if defined(__HIP_DEVICE_COMPILE__)
    // (neg_a, A, neg_b, B, c_mod, C, reuse_a, reuse_b)
    acc = __builtin_amdgcn_wmma_f32_16x16x32_f16(false, a, false, b,
                                                 (short)0, acc, false, false);
#endif
  }

  const int colg  = tcol * 16 + mr;
  const int rbase = trow * 16 + ((lane < 16) ? 0 : 8);
#pragma unroll
  for (int v = 0; v < 8; ++v)
    out[(size_t)(rbase + v) * HD + colg] = acc[v];
}

// ---------------------------------------------------------------------------
// Persistent sequential scan: h_t = tanh(A h_{t-1} + Bx_t), A held in VGPRs.
// d_out arrives holding Bx; step t reads row t-1 (h_{t-1}) and row t (Bx_t),
// overwrites row t with h_t. Cross-WG sync: per-WG monotonic step flags with
// release/acquire semantics through L2 (no central atomic-RMW contention).
// ---------------------------------------------------------------------------
__global__ void __launch_bounds__(SCAN_TPB, 1)
rnn_scan(const float* __restrict__ A_raw,
         const float* __restrict__ h0,
         float* __restrict__ out,
         unsigned int* __restrict__ flags) {
  __shared__ float h_lds[HD];

  const int tid  = threadIdx.x;
  const int lane = tid & 31;
  const int wave = tid >> 5;
  const int g    = blockIdx.x;           // 0..NWG-1
  const int r0   = g * 16 + wave * 2;    // two rows per wave
  const int r1   = r0 + 1;

  // Build A = 0.9*I + 0.1*A_raw on the fly; hold both row slices in VGPRs.
  float a0[32], a1[32];
#pragma unroll
  for (int j = 0; j < 32; ++j) {
    const int k = lane + 32 * j;
    a0[j] = 0.1f * A_raw[(size_t)r0 * HD + k] + ((k == r0) ? 0.9f : 0.0f);
    a1[j] = 0.1f * A_raw[(size_t)r1 * HD + k] + ((k == r1) ? 0.9f : 0.0f);
  }

  for (int t = 0; t < T_STEPS; ++t) {
    // Stage h_{t-1} (or h0 at t=0) into LDS: 256 threads x float4 = 4 KB.
    const float* hsrc = (t == 0) ? h0 : (out + (size_t)(t - 1) * HD);
    *(float4*)(h_lds + tid * 4) = *(const float4*)(hsrc + tid * 4);
    __syncthreads();

    // Prefetch next step's Bx slice for this WG (global_prefetch_b8).
    if (tid == 0 && t + 1 < T_STEPS)
      __builtin_prefetch(out + (size_t)(t + 1) * HD + g * 16, 0, 1);

    // 2-row matvec: 64 fp32 FMAs per lane, conflict-free LDS reads.
    float acc0 = 0.0f, acc1 = 0.0f;
#pragma unroll
    for (int j = 0; j < 32; ++j) {
      const float hv = h_lds[lane + 32 * j];
      acc0 = __builtin_fmaf(a0[j], hv, acc0);
      acc1 = __builtin_fmaf(a1[j], hv, acc1);
    }
    // Wave-wide butterfly reduction (wave32).
#pragma unroll
    for (int off = 16; off > 0; off >>= 1) {
      acc0 += __shfl_xor(acc0, off, 32);
      acc1 += __shfl_xor(acc1, off, 32);
    }

    if (lane == 0) {
      const float bx0 = out[(size_t)t * HD + r0];  // Bx_t (still intact)
      const float bx1 = out[(size_t)t * HD + r1];
      out[(size_t)t * HD + r0] = tanhf(acc0 + bx0);
      out[(size_t)t * HD + r1] = tanhf(acc1 + bx1);
      __threadfence();                              // make h_t device-visible
    }
    __syncthreads();

    // Publish our step, then wait for every WG to reach it.
    if (tid == 0)
      __hip_atomic_store(&flags[g], (unsigned)(t + 1),
                         __ATOMIC_RELEASE, __HIP_MEMORY_SCOPE_AGENT);
    if (tid < NWG) {
      while (__hip_atomic_load(&flags[tid], __ATOMIC_ACQUIRE,
                               __HIP_MEMORY_SCOPE_AGENT) < (unsigned)(t + 1))
        __builtin_amdgcn_s_sleep(1);
    }
    __syncthreads();
  }
}

// ---------------------------------------------------------------------------
// Launch: inputs in setup_inputs() order: x_seq, h0, A_raw, B, c (all fp32).
// ---------------------------------------------------------------------------
extern "C" void kernel_launch(void* const* d_in, const int* in_sizes, int n_in,
                              void* d_out, int out_size, void* d_ws, size_t ws_size,
                              hipStream_t stream) {
  const float* x     = (const float*)d_in[0];
  const float* h0    = (const float*)d_in[1];
  const float* A_raw = (const float*)d_in[2];
  const float* Bm    = (const float*)d_in[3];
  const float* c     = (const float*)d_in[4];
  float* out = (float*)d_out;
  unsigned int* flags = (unsigned int*)d_ws;   // only NWG * 4 bytes used

  rnn_init_flags<<<1, 64, 0, stream>>>(flags);
  rnn_gemm_bx<<<dim3(T_STEPS / 16, (HD / 16) / 8), 256, 0, stream>>>(x, Bm, c, out);
  rnn_scan<<<NWG, SCAN_TPB, 0, stream>>>(A_raw, h0, out, flags);
}
